// LSTMCell_81552839017158
// MI455X (gfx1250) — compile-verified, matrix operand order checked by reading, not run on hardware
//
#include <hip/hip_runtime.h>
#include <hip/hip_bf16.h>
#include <math.h>

typedef __bf16 bf16;
typedef __attribute__((ext_vector_type(16))) __bf16 v16bf;
typedef __attribute__((ext_vector_type(8)))  float  v8f;

#define I_DIM 1024
#define H_DIM 1024
#define K_DIM 2048   // I + H
#define N_DIM 4096   // 4*H
#define BROWS 16
#define WAVES 16
#define THREADS (WAVES * 32)

union AF { v16bf v; float4 f[2]; };   // 32 bytes: one 16-bf16 fragment per lane
union VF { v8f  v; float4 f[2]; };    // 32 bytes: one f32 accumulator tile slice

__device__ __forceinline__ float sigm(float x) { return 1.f / (1.f + __expf(-x)); }

// ---------------- prep: f32 -> bf16 activation matrix Ab[B][2048] = [x | h] ----------------
__global__ void prep_acts(const float* __restrict__ x, const float* __restrict__ h,
                          bf16* __restrict__ Ab, int total) {
    int i = blockIdx.x * blockDim.x + threadIdx.x;
    if (i >= total) return;
    int b = i >> 11, k = i & 2047;
    float v = (k < I_DIM) ? x[(size_t)b * I_DIM + k] : h[(size_t)b * H_DIM + (k - I_DIM)];
    Ab[i] = (bf16)v;
}

// ------- prep: f32 -> bf16 weights, swizzled into WMMA B-fragment-native layout -------
// Wf layout: [kt (64)][nt (256)][lane (32)][elem (16)] bf16  -> 1KB per 32x16 tile,
// so a wave fetches one B fragment as two coalesced b128 loads per lane.
__global__ void prep_weights(const float* __restrict__ Wxh, const float* __restrict__ Whh,
                             bf16* __restrict__ Wf, int total) {
    int i = blockIdx.x * blockDim.x + threadIdx.x;
    if (i >= total) return;
    int k = i >> 12, n = i & 4095;
    float v = (k < I_DIM) ? Wxh[(size_t)k * N_DIM + n] : Whh[(size_t)(k - I_DIM) * N_DIM + n];
    int kt = k >> 5, kk = k & 31, nt = n >> 4, nn = n & 15;
    int lane, idx;
    if      (kk <  8) { lane = nn;      idx = kk;      }
    else if (kk < 16) { lane = nn + 16; idx = kk - 8;  }
    else if (kk < 24) { lane = nn;      idx = kk - 8;  }
    else              { lane = nn + 16; idx = kk - 16; }
    Wf[((size_t)(kt * 256 + nt)) * 512 + lane * 16 + idx] = (bf16)v;
}

// ---------------- fused GEMM + groupLN + LSTM gates + cLN + output ----------------
__global__ __attribute__((amdgpu_flat_work_group_size(THREADS, THREADS),
                          amdgpu_waves_per_eu(4)))
void lstm_fused(const bf16* __restrict__ Ab, const bf16* __restrict__ Wf,
                const float* __restrict__ cprev,
                const float* __restrict__ bias,
                const float* __restrict__ gamma, const float* __restrict__ beta,
                const float* __restrict__ cgam,  const float* __restrict__ cbet,
                float* __restrict__ out_h, float* __restrict__ out_c) {
    __shared__ float lds_g[4][256][16];     // 64KB gate exchange chunk [gate][col][row]
    __shared__ float lds_nc[16][1024];      // 64KB new_c rows
    __shared__ float s_part[4][16][4][2];   // per-(gate,row) partial sums from 4 sub-waves
    __shared__ float s_mean[4][16], s_rstd[4][16];
    __shared__ float s_cm[16], s_cr[16];

    const int tid  = threadIdx.x;
    const int w    = tid >> 5;
    const int lane = tid & 31;
    const int half = lane >> 4;
    const int l16  = lane & 15;
    const int g    = w >> 2;          // gate 0..3 (i,j,f,o)
    const int s    = w & 3;           // 256-col sub-range within the gate
    const int r0   = blockIdx.x * BROWS;
    const int nt0  = w * 16;          // first N-tile (col base = w*256)

    v8f acc[16];
#pragma unroll
    for (int t = 0; t < 16; ++t)
#pragma unroll
        for (int v = 0; v < 8; ++v) acc[t][v] = 0.f;

    const bf16* arow  = Ab + (size_t)(r0 + l16) * K_DIM;
    const bf16* wbase = Wf + (size_t)nt0 * 512 + lane * 16;

    // ---- GEMM: D[16x4096] += A[16x2048] * W[2048x4096], bf16 WMMA, K step 32 ----
    // Double-buffered 4-tile groups; sched_barrier pins the pipeline so register
    // pressure stays bounded (no accumulator spills).
#pragma unroll 1
    for (int kt = 0; kt < 64; ++kt) {
        AF a;
        const bf16* ap = arow + kt * 32 + half * 8;
        a.f[0] = *(const float4*)(ap);
        a.f[1] = *(const float4*)(ap + 16);
        if (kt + 1 < 64)
            __builtin_prefetch((const void*)(arow + (kt + 1) * 32), 0, 1);

        const bf16* bp = wbase + (size_t)kt * (256 * 512);

        AF bb[2][4];
#pragma unroll
        for (int u = 0; u < 4; ++u) {
            const bf16* q = bp + u * 512;
            bb[0][u].f[0] = *(const float4*)(q);
            bb[0][u].f[1] = *(const float4*)(q + 8);
        }
        __builtin_amdgcn_sched_barrier(0);

#pragma unroll
        for (int tt = 0; tt < 4; ++tt) {
            const int cur = tt & 1, nxt = cur ^ 1;
            if (tt < 3) {
#pragma unroll
                for (int u = 0; u < 4; ++u) {
                    const bf16* q = bp + ((tt + 1) * 4 + u) * 512;
                    bb[nxt][u].f[0] = *(const float4*)(q);
                    bb[nxt][u].f[1] = *(const float4*)(q + 8);
                }
            }
#pragma unroll
            for (int u = 0; u < 4; ++u) {
                acc[tt * 4 + u] = __builtin_amdgcn_wmma_f32_16x16x32_bf16(
                    false, a.v, false, bb[cur][u].v, (short)0, acc[tt * 4 + u],
                    false, false);
            }
            __builtin_amdgcn_sched_barrier(0);
        }
    }

    // ---- bias add (LN stats must include bias) ----
#pragma unroll
    for (int t = 0; t < 16; ++t) {
        float bc = bias[g * 1024 + s * 256 + t * 16 + l16];
#pragma unroll
        for (int v = 0; v < 8; ++v) acc[t][v] += bc;
    }

    // ---- per-(gate,row) stats: wave-local partials over 256 cols ----
    float s1[8], s2[8];
#pragma unroll
    for (int v = 0; v < 8; ++v) { s1[v] = 0.f; s2[v] = 0.f; }
#pragma unroll
    for (int t = 0; t < 16; ++t)
#pragma unroll
        for (int v = 0; v < 8; ++v) { float xv = acc[t][v]; s1[v] += xv; s2[v] += xv * xv; }
#pragma unroll
    for (int m = 1; m < 16; m <<= 1)    // reduce within each 16-lane half (rows differ per half)
#pragma unroll
        for (int v = 0; v < 8; ++v) {
            s1[v] += __shfl_xor(s1[v], m, 32);
            s2[v] += __shfl_xor(s2[v], m, 32);
        }
    if (l16 == 0) {
#pragma unroll
        for (int v = 0; v < 8; ++v) {
            s_part[g][v + 8 * half][s][0] = s1[v];
            s_part[g][v + 8 * half][s][1] = s2[v];
        }
    }
    __syncthreads();
    if (tid < 64) {
        int gg = tid >> 4, r = tid & 15;
        float a = 0.f, b = 0.f;
#pragma unroll
        for (int ss = 0; ss < 4; ++ss) { a += s_part[gg][r][ss][0]; b += s_part[gg][r][ss][1]; }
        float mean = a * (1.f / 1024.f);
        float var  = b * (1.f / 1024.f) - mean * mean;
        s_mean[gg][r] = mean;
        s_rstd[gg][r] = rsqrtf(var + 1e-3f);
    }
    __syncthreads();

    // ---- normalize in registers: g*(x-mu)*rstd + b ----
#pragma unroll
    for (int t = 0; t < 16; ++t) {
        int col = g * 1024 + s * 256 + t * 16 + l16;
        float ga = gamma[col], be = beta[col];
#pragma unroll
        for (int v = 0; v < 8; ++v) {
            int m = v + 8 * half;
            acc[t][v] = ga * (acc[t][v] - s_mean[g][m]) * s_rstd[g][m] + be;
        }
    }

    // ---- gate combine in 4 column-chunks of 256 ----
    for (int sc = 0; sc < 4; ++sc) {
        if (s == sc && g < 3) {     // o-gate stays in registers
#pragma unroll
            for (int t = 0; t < 16; ++t) {
                VF u; u.v = acc[t];
                int col = t * 16 + l16;
                *(float4*)&lds_g[g][col][half * 8]     = u.f[0];
                *(float4*)&lds_g[g][col][half * 8 + 4] = u.f[1];
            }
        }
        __syncthreads();
        {
            int r  = tid >> 5;
            int c0 = (tid & 31) * 8;
            float nc[8];
#pragma unroll
            for (int k = 0; k < 8; ++k) {
                int c = c0 + k;
                float iv = lds_g[0][c][r];
                float jv = lds_g[1][c][r];
                float fv = lds_g[2][c][r];
                float cp = cprev[(size_t)(r0 + r) * H_DIM + sc * 256 + c];
                nc[k] = cp * sigm(fv + 1.0f) + sigm(iv) * tanhf(jv);
            }
            float4 f0 = make_float4(nc[0], nc[1], nc[2], nc[3]);
            float4 f1 = make_float4(nc[4], nc[5], nc[6], nc[7]);
            float* lp = &lds_nc[r][sc * 256 + c0];
            *(float4*)(lp)     = f0;
            *(float4*)(lp + 4) = f1;
            float* gp = out_c + (size_t)(r0 + r) * H_DIM + sc * 256 + c0;
            *(float4*)(gp)     = f0;
            *(float4*)(gp + 4) = f1;
        }
        __syncthreads();
    }

    // ---- c layernorm stats: one wave per row ----
    {
        float a = 0.f, b = 0.f;
        const float4* p = (const float4*)&lds_nc[w][lane * 32];
#pragma unroll
        for (int q = 0; q < 8; ++q) {
            float4 f = p[q];
            a += f.x + f.y + f.z + f.w;
            b += f.x * f.x + f.y * f.y + f.z * f.z + f.w * f.w;
        }
#pragma unroll
        for (int m = 1; m < 32; m <<= 1) { a += __shfl_xor(a, m, 32); b += __shfl_xor(b, m, 32); }
        if (lane == 0) {
            float mean = a * (1.f / 1024.f);
            float var  = b * (1.f / 1024.f) - mean * mean;
            s_cm[w] = mean;
            s_cr[w] = rsqrtf(var + 1e-3f);
        }
    }
    __syncthreads();

    // ---- new_h = tanh(cLN) * sigmoid(o): o-gate waves still hold normalized o ----
    if (g == 3) {
#pragma unroll
        for (int t = 0; t < 16; ++t) {
            int col = s * 256 + t * 16 + l16;
            float gc = cgam[col], bc = cbet[col];
#pragma unroll
            for (int v = 0; v < 8; ++v) {
                int m = v + 8 * half;
                float cn = (lds_nc[m][col] - s_cm[m]) * s_cr[m];
                float hv = tanhf(gc * cn + bc) * sigm(acc[t][v]);
                out_h[(size_t)(r0 + m) * H_DIM + col] = hv;
            }
        }
    }
}

extern "C" void kernel_launch(void* const* d_in, const int* in_sizes, int n_in,
                              void* d_out, int out_size, void* d_ws, size_t ws_size,
                              hipStream_t stream) {
    const float* x    = (const float*)d_in[0];
    const float* c    = (const float*)d_in[1];
    const float* h    = (const float*)d_in[2];
    const float* Wxh  = (const float*)d_in[3];
    const float* Whh  = (const float*)d_in[4];
    const float* bias = (const float*)d_in[5];
    const float* gam  = (const float*)d_in[6];
    const float* bet  = (const float*)d_in[7];
    const float* cgam = (const float*)d_in[8];
    const float* cbet = (const float*)d_in[9];

    int B = in_sizes[1] / H_DIM;                      // c is [B, H]
    bf16* Wf = (bf16*)d_ws;                           // 2048*4096*2 = 16 MiB
    bf16* Ab = (bf16*)((char*)d_ws + (size_t)K_DIM * N_DIM * 2);  // B*2048*2 = 32 MiB

    float* out_h = (float*)d_out;
    float* out_c = out_h + (size_t)B * H_DIM;

    int wtot = K_DIM * N_DIM;
    prep_weights<<<(wtot + 255) / 256, 256, 0, stream>>>(Wxh, Whh, Wf, wtot);
    int atot = B * K_DIM;
    prep_acts<<<(atot + 255) / 256, 256, 0, stream>>>(x, h, Ab, atot);
    lstm_fused<<<B / BROWS, THREADS, 0, stream>>>(Ab, Wf, c, bias, gam, bet, cgam, cbet,
                                                  out_h, out_c);
}
